// RN_37752762532001
// MI455X (gfx1250) — matrix operand order repaired and checked by reading, not compiled, not run on hardware
//
#include <hip/hip_runtime.h>

typedef __attribute__((ext_vector_type(16))) _Float16 v16h;
typedef __attribute__((ext_vector_type(8)))  _Float16 v8h;
typedef __attribute__((ext_vector_type(4)))  _Float16 v4h;
typedef __attribute__((ext_vector_type(8)))  float    v8f;

union V16 { v16h v; v8h h[2]; };

#define N_OBJ   64
#define N_PAIR  2016
#define N_TILE  126      // 2016 pairs / 16 per tile
#define BATCH   1024

// ---------------- workspace layout (bytes) ----------------
// [    0,  8192)  w0a f16, packed in WMMA B-operand order
// [ 8192, 16384)  w0b f16, packed
// [16384, 24576)  w1  f16, packed
// [24576, 32768)  pij int32: i | (j<<16)   (2016 used)
// [32768, 40960)  pw  f32                  (2016 used)
//
// B-operand packing: entry ((nt*2+chunk)*32 + lane) is a v16h whose element j
// equals W[(nt*16 + lane%16)*64 + chunk*32 + (j&8)*2 + (lane/16)*8 + (j&7)]   (W row-major [n][k])

__global__ __launch_bounds__(256) void rn_prep_kernel(
    const float* __restrict__ w0a, const float* __restrict__ w0b,
    const float* __restrict__ w1,
    _Float16* __restrict__ w0ap, _Float16* __restrict__ w0bp,
    _Float16* __restrict__ w1p,
    int* __restrict__ pij, float* __restrict__ pwf)
{
  int t = blockIdx.x * 256 + threadIdx.x;          // 2048 threads
  for (int o = t; o < 64 * 64; o += 2048) {
    const int j    = o & 15;
    const int q    = o >> 4;
    const int lane = q & 31;
    const int ch   = (q >> 5) & 1;
    const int nt   = q >> 6;
    const int src  = (nt * 16 + (lane & 15)) * 64 + ch * 32 + (j & 8) * 2 + (lane >> 4) * 8 + (j & 7);
    w0ap[o] = (_Float16)w0a[src];
    w0bp[o] = (_Float16)w0b[src];
    w1p[o]  = (_Float16)w1[src];
  }
  if (t < N_PAIR) {
    // pairs enumerated by gap d=1..63, then i=0..63-d
    int d = 1, rem = t;
    while (rem >= N_OBJ - d) { rem -= N_OBJ - d; ++d; }
    pij[t] = rem | ((rem + d) << 16);
    pwf[t] = 1.0f / ((float)(N_OBJ - 1) * (float)(N_OBJ - d));
  }
}

__global__ __launch_bounds__(256) void rn_main_kernel(
    const float* __restrict__ x,
    const float* __restrict__ b0g,  const float* __restrict__ b1g,
    const float* __restrict__ wm1,  const float* __restrict__ bm1,
    const float* __restrict__ wm2,  const float* __restrict__ bm2,
    const _Float16* __restrict__ w0ap, const _Float16* __restrict__ w0bp,
    const _Float16* __restrict__ w1p,
    const int* __restrict__ pij, const float* __restrict__ pwf,
    float* __restrict__ out)
{
  __shared__ _Float16 xh [64 * 64];   // 8 KB  x  (f16 row-major [obj][feat])
  __shared__ _Float16 Afh[64 * 64];   // 8 KB  A' = xo*w0a^T + b0/2  (f16)
  __shared__ _Float16 Bfh[64 * 64];   // 8 KB  B' = xo*w0b^T + b0/2  (f16)
  __shared__ float    b0s[64], b1s[64], pooled[64], zs[64];

  const int tid  = threadIdx.x;
  const int wave = tid >> 5;
  const int lane = tid & 31;
  const int g    = lane >> 4;   // K-half (A operand) / M-half (C operand)
  const int m    = lane & 15;   // M row  (A operand) / N col  (C operand)

  // ---- stage x into LDS as f16 (vector loads/stores), biases, pooled=0 ----
  const float4* xb4 = (const float4*)(x + (size_t)blockIdx.x * (64 * 64));
  for (int k = tid; k < 1024; k += 256) {
    const float4 f = xb4[k];
    v4h h;
    h[0] = (_Float16)f.x; h[1] = (_Float16)f.y;
    h[2] = (_Float16)f.z; h[3] = (_Float16)f.w;
    *(v4h*)&xh[k * 4] = h;                       // ds_store_b64
  }
  if (tid < 64) { b0s[tid] = b0g[tid]; b1s[tid] = b1g[tid]; pooled[tid] = 0.0f; }
  __syncthreads();

  // =========== Stage A: A' = xo*w0a^T + b0/2, B' likewise, via WMMA ========
  // 32 tile-jobs: job = mat*16 + mtile*4 + ntile ; wave w owns jobs 4w..4w+3
  const v16h* w0apv = (const v16h*)w0ap;
  const v16h* w0bpv = (const v16h*)w0bp;
  for (int job = wave * 4; job < wave * 4 + 4; ++job) {
    const int mat   = job >> 4;
    const int mtile = (job >> 2) & 3;
    const int ntile = job & 3;
    const v16h* wv  = mat ? w0bpv : w0apv;

    const int row = mtile * 16 + m;
    V16 a0, a1, wb0, wb1;
    a0.h[0] = *(const v8h*)&xh[row * 64 + g * 8];        // K = g8+0..7
    a0.h[1] = *(const v8h*)&xh[row * 64 + 16 + g * 8];   // K = 16+g8..
    a1.h[0] = *(const v8h*)&xh[row * 64 + 32 + g * 8];
    a1.h[1] = *(const v8h*)&xh[row * 64 + 48 + g * 8];
    wb0.v = wv[(ntile * 2 + 0) * 32 + lane];             // packed B-operand
    wb1.v = wv[(ntile * 2 + 1) * 32 + lane];

    v8f acc = {};
    acc = __builtin_amdgcn_wmma_f32_16x16x32_f16(false, a0.v, false, wb0.v,
                                                 (short)0, acc, false, false);
    acc = __builtin_amdgcn_wmma_f32_16x16x32_f16(false, a1.v, false, wb1.v,
                                                 (short)0, acc, false, false);

    _Float16* dst = mat ? Bfh : Afh;
    const float hb = 0.5f * b0s[ntile * 16 + m];         // fold bias here
    #pragma unroll
    for (int r = 0; r < 8; ++r) {                        // C: M = r+8g, N = m
      const int M = mtile * 16 + g * 8 + r;
      dst[M * 64 + ntile * 16 + m] = (_Float16)(acc[r] + hb);
    }
  }
  __syncthreads();

  // =========== Stage B: per-pair layer 1 + weighted pooling ================
  const v16h* w1pv = (const v16h*)w1p;
  V16 w1b[4][2];
  #pragma unroll
  for (int nt = 0; nt < 4; ++nt) {
    w1b[nt][0].v = w1pv[(nt * 2 + 0) * 32 + lane];
    w1b[nt][1].v = w1pv[(nt * 2 + 1) * 32 + lane];
  }

  // bias as the C initializer, hoisted out of the tile loop (SSA, no copies)
  v8f cbias[4];
  #pragma unroll
  for (int nt = 0; nt < 4; ++nt) {
    const float bv = b1s[nt * 16 + m];                   // C layout: N = m
    #pragma unroll
    for (int r = 0; r < 8; ++r) cbias[nt][r] = bv;
  }

  float accp[4] = {0.f, 0.f, 0.f, 0.f};
  const v8h z8 = {};

  for (int tile = wave; tile < N_TILE; tile += 8) {
    const int pb = tile * 16;
    const int pv = pij[pb + m];
    const int ip = pv & 0xffff;
    const int jp = pv >> 16;

    // D = relu(A'[ip] + B'[jp]) built with packed-f16 math, 8x ds_load_b128
    const v8h* pa = (const v8h*)&Afh[ip * 64 + g * 8];
    const v8h* pq = (const v8h*)&Bfh[jp * 64 + g * 8];
    V16 a0, a1;
    a0.h[0] = __builtin_elementwise_max(pa[0] + pq[0], z8);  // K g8..
    a0.h[1] = __builtin_elementwise_max(pa[2] + pq[2], z8);  // K 16+g8..
    a1.h[0] = __builtin_elementwise_max(pa[4] + pq[4], z8);  // K 32+g8..
    a1.h[1] = __builtin_elementwise_max(pa[6] + pq[6], z8);  // K 48+g8..

    const float4* pw4 = (const float4*)&pwf[pb + g * 8];
    const float4 wlo = pw4[0], whi = pw4[1];
    float wp[8] = {wlo.x, wlo.y, wlo.z, wlo.w, whi.x, whi.y, whi.z, whi.w};

    // issue all 8 WMMAs into 4 independent accumulators first...
    v8f cc[4];
    #pragma unroll
    for (int nt = 0; nt < 4; ++nt) {
      cc[nt] = __builtin_amdgcn_wmma_f32_16x16x32_f16(false, a0.v, false,
                                                      w1b[nt][0].v, (short)0,
                                                      cbias[nt], false, false);
      cc[nt] = __builtin_amdgcn_wmma_f32_16x16x32_f16(false, a1.v, false,
                                                      w1b[nt][1].v, (short)0,
                                                      cc[nt], false, false);
    }
    // ...then consume: pooling VALU overlaps later WMMAs / hazard windows
    #pragma unroll
    for (int nt = 0; nt < 4; ++nt) {
      #pragma unroll
      for (int r = 0; r < 8; ++r)            // C layout: pair = pb + 8g + r
        accp[nt] += wp[r] * fmaxf(cc[nt][r], 0.0f);
    }
  }
  #pragma unroll
  for (int nt = 0; nt < 4; ++nt)
    atomicAdd(&pooled[nt * 16 + m], accp[nt]);   // ds_add_f32 reduce
  __syncthreads();

  // =========== Stage C: pooled -> 64 -> 10 MLP =============================
  if (tid < 64) {
    float a = bm1[tid];
    for (int c = 0; c < 64; ++c) a += pooled[c] * wm1[tid * 64 + c];
    zs[tid] = fmaxf(a, 0.0f);
  }
  __syncthreads();
  if (tid < 10) {
    float o = bm2[tid];
    for (int c = 0; c < 64; ++c) o += zs[c] * wm2[tid * 64 + c];
    out[(size_t)blockIdx.x * 10 + tid] = o;
  }
}

extern "C" void kernel_launch(void* const* d_in, const int* in_sizes, int n_in,
                              void* d_out, int out_size, void* d_ws, size_t ws_size,
                              hipStream_t stream)
{
  const float* x   = (const float*)d_in[0];
  const float* w0a = (const float*)d_in[1];
  const float* w0b = (const float*)d_in[2];
  const float* b0  = (const float*)d_in[3];
  const float* w1  = (const float*)d_in[4];
  const float* b1  = (const float*)d_in[5];
  const float* wm1 = (const float*)d_in[6];
  const float* bm1 = (const float*)d_in[7];
  const float* wm2 = (const float*)d_in[8];
  const float* bm2 = (const float*)d_in[9];

  char* ws = (char*)d_ws;
  _Float16* w0ap = (_Float16*)(ws + 0);
  _Float16* w0bp = (_Float16*)(ws + 8192);
  _Float16* w1p  = (_Float16*)(ws + 16384);
  int*      pij  = (int*)(ws + 24576);
  float*    pwf  = (float*)(ws + 32768);

  rn_prep_kernel<<<8, 256, 0, stream>>>(w0a, w0b, w1, w0ap, w0bp, w1p, pij, pwf);
  rn_main_kernel<<<BATCH, 256, 0, stream>>>(x, b0, b1, wm1, bm1, wm2, bm2,
                                            w0ap, w0bp, w1p, pij, pwf,
                                            (float*)d_out);
}
